// QRNN_30442728194657
// MI455X (gfx1250) — matrix operand order repaired and checked by reading, not compiled, not run on hardware
//
#include <hip/hip_runtime.h>
#include <hip/hip_bf16.h>
#include <math.h>

// ---------------------------------------------------------------------------
// QRNN forward for MI455X (gfx1250, wave32, WMMA).
//   Phase 1: gates = act(x @ W^T + b)   -- bf16 WMMA GEMM, fused activations
//   Phase 2: c_t = f*c + (1-f)*z ; out = o*c   -- parallel over B*H channels
// ---------------------------------------------------------------------------

typedef __attribute__((ext_vector_type(16))) __bf16 v16bf;
typedef __attribute__((ext_vector_type(8)))  __bf16 v8bf;
typedef __attribute__((ext_vector_type(4)))  __bf16 v4bf;
typedef __attribute__((ext_vector_type(8)))  float  v8f;

#define QB   32
#define QS   512
#define QI   512      // K
#define QH   1024
#define QG3  3072     // 3*H  (N)
#define QM   16384    // B*S  (M)

#define LDSP 272      // padded LDS row pitch (bf16 elems): 272*2B = 544B

// Raw (fp32) A-fragment data: 16 floats per lane, loaded as 4x float4.
struct RawA { float4 a0, a1, a2, a3; };

// Documented 16-bit A layout (16x32):
//   lanes 0-15 : K = kg+0..7  and kg+16..23
//   lanes 16-31: K = kg+8..15 and kg+24..31
static __device__ inline RawA load_raw_a(const float* __restrict__ aRow,
                                         int kg, int half) {
    const float4* p0 = reinterpret_cast<const float4*>(aRow + kg + half * 8);
    const float4* p1 = reinterpret_cast<const float4*>(aRow + kg + 16 + half * 8);
    RawA r;
    r.a0 = p0[0]; r.a1 = p0[1];
    r.a2 = p1[0]; r.a3 = p1[1];
    return r;
}

static __device__ inline v16bf cvt_a(const RawA& r) {
    v16bf av;
    av[0]  = (__bf16)r.a0.x; av[1]  = (__bf16)r.a0.y;
    av[2]  = (__bf16)r.a0.z; av[3]  = (__bf16)r.a0.w;
    av[4]  = (__bf16)r.a1.x; av[5]  = (__bf16)r.a1.y;
    av[6]  = (__bf16)r.a1.z; av[7]  = (__bf16)r.a1.w;
    av[8]  = (__bf16)r.a2.x; av[9]  = (__bf16)r.a2.y;
    av[10] = (__bf16)r.a2.z; av[11] = (__bf16)r.a2.w;
    av[12] = (__bf16)r.a3.x; av[13] = (__bf16)r.a3.y;
    av[14] = (__bf16)r.a3.z; av[15] = (__bf16)r.a3.w;
    return av;
}

// ---------------------------------------------------------------------------
// GEMM: G[m, g] = act( sum_k x[m,k] * W[g,k] + b[g] )
// block = 256 threads (8 waves). Block tile: 128 (M) x 64 (N).
// Each wave: 16 (M) x 64 (N) via 4x v_wmma_f32_16x16x32_bf16 per K-step.
// ---------------------------------------------------------------------------
__global__ __launch_bounds__(256)
void qrnn_gemm_act(const float* __restrict__ X,      // [M, I]
                   const float* __restrict__ W,      // [3H, I]
                   const float* __restrict__ bias,   // [3H]
                   float* __restrict__ G)            // [M, 3H] activated gates
{
    __shared__ __attribute__((aligned(16))) __bf16 lw[64 * LDSP];

    const int tid  = threadIdx.x;
    const int wave = tid >> 5;
    const int lane = tid & 31;
    const int lrow = lane & 15;   // M row (A/C) or N col (B) within tile
    const int half = lane >> 4;   // selects K sub-block / C row group

    const int n0 = blockIdx.x * 64;               // N tile base (3072/64 = 48)
    const int m0 = blockIdx.y * 128 + wave * 16;  // M tile base (16384/128 = 128)

    const float* aRow = X + (size_t)(m0 + lrow) * QI;

    v8f acc[4];
#pragma unroll
    for (int j = 0; j < 4; ++j)
#pragma unroll
        for (int r = 0; r < 8; ++r) acc[j][r] = 0.0f;

    // Depth-2 raw A prefetch pipeline: loads issued at step i-2 are consumed
    // (converted) at step i, loads for step i+2 issued right after, so the
    // loadcnt wait lands ~2 k-steps (8 WMMAs + 16 DS loads) downstream.
    RawA rawA[2];
    rawA[0] = load_raw_a(aRow, 0, half);
    rawA[1] = load_raw_a(aRow, 32, half);
    // Parity-alternating converted fragments: converts of step i+1 never
    // write registers read by step i's in-flight WMMAs (kills WAR v_nops).
    v16bf avbuf[2];

#pragma unroll
    for (int kc = 0; kc < QI; kc += 256) {
        __syncthreads();  // protect LDS reads of previous chunk
        // Cooperative stage of W[n0..n0+63][kc..kc+255] into LDS as bf16.
        {
            const int rsub = tid >> 6;         // 0..3
            const int col  = (tid & 63) * 4;   // 0..252
#pragma unroll
            for (int rr = 0; rr < 16; ++rr) {
                const int row = rr * 4 + rsub;
                float4 v = *reinterpret_cast<const float4*>(
                    W + (size_t)(n0 + row) * QI + kc + col);
                v4bf p;
                p[0] = (__bf16)v.x; p[1] = (__bf16)v.y;
                p[2] = (__bf16)v.z; p[3] = (__bf16)v.w;
                *reinterpret_cast<v4bf*>(&lw[row * LDSP + col]) = p;
            }
        }
        __syncthreads();

#pragma unroll
        for (int kk = 0; kk < 256; kk += 32) {
            const int step = (kc + kk) >> 5;   // 0..15, compile-time constant
            const int pb   = step & 1;

            // Consume raw data loaded 2 steps ago; convert into the parity
            // slot, then immediately re-issue loads for step+2.
            avbuf[pb] = cvt_a(rawA[pb]);
            const int kp = kc + kk + 64;
            if (kp < QI) rawA[pb] = load_raw_a(aRow, kp, half);

            // Load ALL B fragments first (DS clause), then 4 WMMAs
            // back-to-back (distinct accumulators, shared sources).
            // B layout (32x16 bf16): lane = N column, lane half h: K = 8h..8h+7 and 16+8h..16+8h+7.
            v16bf bv[4];
#pragma unroll
            for (int j = 0; j < 4; ++j) {
                const v8bf* bp = reinterpret_cast<const v8bf*>(
                    &lw[(j * 16 + lrow) * LDSP + kk + half * 8]);
                v8bf blo = bp[0], bhi = bp[2];
#pragma unroll
                for (int i = 0; i < 8; ++i) { bv[j][i] = blo[i]; bv[j][i + 8] = bhi[i]; }
            }
#pragma unroll
            for (int j = 0; j < 4; ++j) {
                acc[j] = __builtin_amdgcn_wmma_f32_16x16x32_bf16(
                    false, avbuf[pb], false, bv[j], (short)0, acc[j], false, false);
            }
        }
    }

    // ---- Epilogue: bias + gate activation, store to G.
    // C layout: VGPR r -> M = m0 + half*8 + r ; N = n0 + j*16 + lrow
    const int gr0 = m0 + half * 8;
#pragma unroll
    for (int j = 0; j < 4; ++j) {
        const int gcol = n0 + j * 16 + lrow;
        const float bb = bias[gcol];
        const bool is_tanh = (gcol < QH);   // gate region uniform per subtile
#pragma unroll
        for (int r = 0; r < 8; ++r) {
            float v = acc[j][r] + bb;
            float a = is_tanh ? tanhf(v)
                              : 1.0f / (1.0f + __expf(-v));
            G[(size_t)(gr0 + r) * QG3 + gcol] = a;
        }
    }
}

// ---------------------------------------------------------------------------
// Scan: one thread per (b, h) channel; sequential over S.
//   c = f*c + (1-f)*z = fma(f, c-z, z);  out[b,s,h] = o*c
// Loads/stores coalesced across h.
// ---------------------------------------------------------------------------
__global__ __launch_bounds__(256)
void qrnn_scan(const float* __restrict__ G,      // [M, 3H] activated gates
               const float* __restrict__ c0,     // [B, H]
               float* __restrict__ out,          // [B, S, H]
               float* __restrict__ clast)        // [B, H]
{
    const int t = blockIdx.x * 256 + threadIdx.x;   // 0 .. B*H-1
    const int b = t >> 10;                          // /H
    const int h = t & (QH - 1);

    float c = c0[t];
    size_t base  = (size_t)b * QS * QG3 + h;        // G[b*S + s][.. + h]
    size_t obase = (size_t)b * QS * QH + h;

#pragma unroll 4
    for (int s = 0; s < QS; ++s) {
        // speculative prefetch a few timesteps ahead (global_prefetch_b8)
        __builtin_prefetch(G + base + (size_t)8 * QG3, 0, 1);
        const float z = G[base];
        const float f = G[base + QH];
        const float o = G[base + 2 * QH];
        c = fmaf(f, c - z, z);
        out[obase] = o * c;
        base  += QG3;
        obase += QH;
    }
    clast[t] = c;
}

// ---------------------------------------------------------------------------
extern "C" void kernel_launch(void* const* d_in, const int* in_sizes, int n_in,
                              void* d_out, int out_size, void* d_ws, size_t ws_size,
                              hipStream_t stream) {
    const float* x = (const float*)d_in[0];   // [B, S, I]
    const float* h = (const float*)d_in[1];   // [1, B, H]
    const float* W = (const float*)d_in[2];   // [3H, I]
    const float* b = (const float*)d_in[3];   // [3H]

    float* out   = (float*)d_out;                       // [B, S, H]
    float* clast = out + (size_t)QB * QS * QH;          // [1, B, H]
    float* G     = (float*)d_ws;                        // [M, 3H] = 192 MiB

    dim3 gemm_grid(QG3 / 64, QM / 128);                 // (48, 128)
    qrnn_gemm_act<<<gemm_grid, 256, 0, stream>>>(x, W, b, G);

    qrnn_scan<<<(QB * QH) / 256, 256, 0, stream>>>(G, h, out, clast);
}
